// STORN_71786083385629
// MI455X (gfx1250) — matrix operand order repaired and checked by reading, not compiled, not run on hardware
//
#include <hip/hip_runtime.h>
#include <cstdint>

// STORN / TD-VAE pipeline for MI455X (gfx1250, wave32, WMMA).
// - All GEMMs: v_wmma_f32_16x16x32_f16 with inline-0 C; bias folded into the
//   elementwise epilogues.
// - Internal ws tensors (proj, dseq) fragment-native: [rowblock16][tile][lane][8h].
// - GRU scans: 64 persistent single-wave chains; W_hh register-resident
//   (24 B-fragments); per-step input projection DMA'd one step ahead via
//   tensor_load_to_lds (TDM) double-buffered in LDS, synced by s_wait_tensorcnt.

#define L_SEQ 512
#define B_SZ  1024
#define LB    (L_SEQ * B_SZ)
#define NRB   (B_SZ / 16)   // rowblocks per timestep = 64

typedef __attribute__((ext_vector_type(16))) _Float16 v16h;
typedef __attribute__((ext_vector_type(8)))  _Float16 v8h;
typedef __attribute__((ext_vector_type(8)))  float    v8f;
typedef __attribute__((ext_vector_type(4)))  unsigned u32x4;
typedef __attribute__((ext_vector_type(8)))  unsigned u32x8;

#define LDS_FENCE() asm volatile("s_wait_dscnt 0" ::: "memory")

__device__ __forceinline__ unsigned lane_id() { return threadIdx.x & 31u; }
__device__ __forceinline__ unsigned c_col()   { return lane_id() & 15u; }
__device__ __forceinline__ unsigned c_m0()    { return (lane_id() & 16u) ? 8u : 0u; }

__device__ __forceinline__ float reluf(float x) { return x > 0.f ? x : 0.f; }
__device__ __forceinline__ float sigf(float x)  { return 1.f / (1.f + __expf(-x)); }

__device__ __forceinline__ v8f wmma16(v16h a, v16h b, v8f c) {
  return __builtin_amdgcn_wmma_f32_16x16x32_f16(false, a, false, b, (short)0, c,
                                                false, false);
}

// ---- A-fragment (16x32 f16): lane<16 row M=lane K{0..7,16..23};
//      lane>=16 row M=lane-16 K{8..15,24..31}; two contiguous 16B runs.
__device__ __forceinline__ v16h a_load_h(const _Float16* src, int ld) {
  unsigned l = lane_id();
  const _Float16* p = src + (size_t)(l & 15u) * (unsigned)ld + ((l & 16u) ? 8u : 0u);
  v16h r;
#pragma unroll
  for (int i = 0; i < 8; ++i) { r[i] = p[i]; r[i + 8] = p[i + 16]; }
  return r;
}
__device__ __forceinline__ v16h a_load_f(const float* src, int ld) {
  unsigned l = lane_id();
  const float* p = src + (size_t)(l & 15u) * (unsigned)ld + ((l & 16u) ? 8u : 0u);
  v16h r;
#pragma unroll
  for (int i = 0; i < 8; ++i) { r[i] = (_Float16)p[i]; r[i + 8] = (_Float16)p[i + 16]; }
  return r;
}

// ---- B-fragment (32x16) from row-major W[N_out][K_in] f16.
__device__ __forceinline__ v16h b_load(const _Float16* W, int ldk, int n0, int k0) {
  unsigned l = lane_id();
  const _Float16* p = W + (size_t)(n0 + (l & 15u)) * (unsigned)ldk + (unsigned)k0 +
                      ((l & 16u) ? 8u : 0u);
  v16h r;
#pragma unroll
  for (int i = 0; i < 8; ++i) { r[i] = p[i]; r[i + 8] = p[i + 16]; }
  return r;
}

// ---- C/D fragment helpers (element (m = i + 8*(lane>=16), n = lane&15))
__device__ __forceinline__ v8f c_load_f(const float* p, int ld) {
  unsigned m0 = c_m0(), col = c_col();
  v8f c;
#pragma unroll
  for (int i = 0; i < 8; ++i) c[i] = p[(size_t)(m0 + i) * (unsigned)ld + col];
  return c;
}
__device__ __forceinline__ void c_store_f(float* p, int ld, v8f c) {
  unsigned m0 = c_m0(), col = c_col();
#pragma unroll
  for (int i = 0; i < 8; ++i) p[(size_t)(m0 + i) * (unsigned)ld + col] = c[i];
}
__device__ __forceinline__ void c_store_h(_Float16* p, int ld, v8f c) {
  unsigned m0 = c_m0(), col = c_col();
#pragma unroll
  for (int i = 0; i < 8; ++i) p[(size_t)(m0 + i) * (unsigned)ld + col] = (_Float16)c[i];
}

// ---- fragment-native C-tile I/O: tile = [32 lanes][8 halves] = 512 bytes.
__device__ __forceinline__ void cn_store_b(_Float16* tile, v8f c, float bias) {
  v8h h;
#pragma unroll
  for (int i = 0; i < 8; ++i) h[i] = (_Float16)(c[i] + bias);
  *(v8h*)(tile + lane_id() * 8) = h;   // one b128 per lane
}
__device__ __forceinline__ void cn_store(_Float16* tile, v8f c) {
  v8h h;
#pragma unroll
  for (int i = 0; i < 8; ++i) h[i] = (_Float16)c[i];
  *(v8h*)(tile + lane_id() * 8) = h;
}
__device__ __forceinline__ v8f cn_load(const _Float16* tile) {
  v8h h = *(const v8h*)(tile + lane_id() * 8);
  v8f c;
#pragma unroll
  for (int i = 0; i < 8; ++i) c[i] = (float)h[i];
  return c;
}
__device__ __forceinline__ v8h cn_load_raw(const _Float16* tile) {
  return *(const v8h*)(tile + lane_id() * 8);
}

// ---- TDM: 1-D tile copy global -> LDS (contiguous, 8B elements).
__device__ __forceinline__ void tdm_fetch(const _Float16* gsrc, _Float16* ldst,
                                          unsigned n8) {
  uint64_t ga = (uint64_t)(uintptr_t)gsrc;
  unsigned la = (unsigned)(uintptr_t)ldst;
  u32x4 g0;
  g0[0] = 1u;                                                // count = 1
  g0[1] = la;                                                // lds_addr
  g0[2] = (unsigned)ga;                                      // global_addr[31:0]
  g0[3] = ((unsigned)(ga >> 32) & 0x01FFFFFFu) | (2u << 30); // [56:32] | type=2
  u32x8 g1;
  g1[0] = 3u << 16;                        // data_size = 8 bytes
  g1[1] = (n8 & 0xFFFFu) << 16;            // tensor_dim0 lo16
  g1[2] = (n8 >> 16) | (1u << 16);         // tensor_dim0 hi | tensor_dim1 = 1
  g1[3] = n8 << 16;                        // tile_dim0
  g1[4] = 0u;                              // tile_dim1 = tile_dim2 = 0
  g1[5] = n8;                              // tensor_dim0_stride lo32
  g1[6] = 0u;
  g1[7] = 0u;
  asm volatile("tensor_load_to_lds %0, %1" :: "s"(g0), "s"(g1) : "memory");
}

// ---------------- weight fp32 -> fp16 prep ----------------
struct PrepSeg { const float* src; int n; int dstOff; };
struct PrepArgs { PrepSeg seg[14]; };

__global__ void prep_kernel(PrepArgs a, _Float16* __restrict__ dst) {
  int tid = blockIdx.x * blockDim.x + threadIdx.x;
  int stride = gridDim.x * blockDim.x;
  for (int s = 0; s < 14; ++s)
    for (int i = tid; i < a.seg[s].n; i += stride)
      dst[a.seg[s].dstOff + i] = (_Float16)a.seg[s].src[i];
}

// f16 weight pool offsets (halves)
#define WQ_X1   0
#define WQ_X2   2048
#define WQ_U1   6144
#define WQ_U2   8192
#define WQ_Z1   12288
#define WQ_Z2   14336
#define WQ_IIH  18432
#define WQ_IHH  30720
#define WQ_P1   43008
#define WQ_P2   51200
#define WQ_PMU  59392
#define WQ_PLS  63488
#define WQ_GIH  67584
#define WQ_GHH  92160

// ---------------- fused embed: x_embed, u_embed, inf input projection --------
__global__ __launch_bounds__(256) void embed_kernel(
    const float* __restrict__ xin, const float* __restrict__ uin,
    const _Float16* __restrict__ wq,
    const float* __restrict__ bx1, const float* __restrict__ bx2,
    const float* __restrict__ bu1, const float* __restrict__ bu2,
    const float* __restrict__ binf,
    float* __restrict__ xemb_out, _Float16* __restrict__ uemb,
    _Float16* __restrict__ proj) {
  __shared__ __align__(16) _Float16 lds[8][16 * 72];
  unsigned wid = threadIdx.x >> 5;
  _Float16* sm = lds[wid];
  size_t rbg = (size_t)blockIdx.x * 8u + wid;   // global rowblock id
  size_t r0 = rbg * 16u;
  unsigned m0 = c_m0(), col = c_col();

  // per-lane bias values (column-indexed; broadcast over rows in epilogues)
  float bvx1[4], bvx2[4], bvu1[4], bvu2[4], bvp[12];
#pragma unroll
  for (int j = 0; j < 4; ++j) {
    bvx1[j] = bx1[j * 16 + col]; bvx2[j] = bx2[j * 16 + col];
    bvu1[j] = bu1[j * 16 + col]; bvu2[j] = bu2[j * 16 + col];
  }
#pragma unroll
  for (int j = 0; j < 12; ++j) bvp[j] = binf[j * 16 + col];

  // ---- observations path ----
  {
    v16h a = a_load_f(xin + r0 * 32, 32);
    v16h w1[4];
#pragma unroll
    for (int j = 0; j < 4; ++j) w1[j] = b_load(wq + WQ_X1, 32, j * 16, 0);
    v8f acc1[4];
#pragma unroll
    for (int j = 0; j < 4; ++j) { v8f z = {}; acc1[j] = wmma16(a, w1[j], z); }
#pragma unroll
    for (int j = 0; j < 4; ++j)
#pragma unroll
      for (int i = 0; i < 8; ++i)
        sm[(m0 + i) * 72 + j * 16 + col] = (_Float16)reluf(acc1[j][i] + bvx1[j]);
    LDS_FENCE();
    v16h h0 = a_load_h(sm, 72), h1 = a_load_h(sm + 32, 72);

    v16h w2[4][2];
#pragma unroll
    for (int j = 0; j < 4; ++j) {
      w2[j][0] = b_load(wq + WQ_X2, 64, j * 16, 0);
      w2[j][1] = b_load(wq + WQ_X2, 64, j * 16, 32);
    }
    v8f acc2[4];
#pragma unroll
    for (int j = 0; j < 4; ++j) { v8f z = {}; acc2[j] = wmma16(h0, w2[j][0], z); }
#pragma unroll
    for (int j = 0; j < 4; ++j) acc2[j] = wmma16(h1, w2[j][1], acc2[j]);
#pragma unroll
    for (int j = 0; j < 4; ++j) {
#pragma unroll
      for (int i = 0; i < 8; ++i) acc2[j][i] = reluf(acc2[j][i] + bvx2[j]);
      c_store_f(xemb_out + r0 * 64 + j * 16, 64, acc2[j]);
#pragma unroll
      for (int i = 0; i < 8; ++i)
        sm[(m0 + i) * 72 + j * 16 + col] = (_Float16)acc2[j][i];
    }
    LDS_FENCE();
    v16h xa0 = a_load_h(sm, 72), xa1 = a_load_h(sm + 32, 72);

    // inference input projection: 64 -> 192 (3 groups of 4 tiles)
#pragma unroll
    for (int g = 0; g < 3; ++g) {
      v16h bw[4][2];
#pragma unroll
      for (int q = 0; q < 4; ++q) {
        bw[q][0] = b_load(wq + WQ_IIH, 64, (g * 4 + q) * 16, 0);
        bw[q][1] = b_load(wq + WQ_IIH, 64, (g * 4 + q) * 16, 32);
      }
      v8f acc[4];
#pragma unroll
      for (int q = 0; q < 4; ++q) { v8f z = {}; acc[q] = wmma16(xa0, bw[q][0], z); }
#pragma unroll
      for (int q = 0; q < 4; ++q) acc[q] = wmma16(xa1, bw[q][1], acc[q]);
#pragma unroll
      for (int q = 0; q < 4; ++q)
        cn_store_b(proj + rbg * 3072 + (g * 4 + q) * 256, acc[q], bvp[g * 4 + q]);
    }
  }
  // ---- external input path ----
  {
    v16h a = a_load_f(uin + r0 * 32, 32);
    v16h w1[4];
#pragma unroll
    for (int j = 0; j < 4; ++j) w1[j] = b_load(wq + WQ_U1, 32, j * 16, 0);
    v8f acc1[4];
#pragma unroll
    for (int j = 0; j < 4; ++j) { v8f z = {}; acc1[j] = wmma16(a, w1[j], z); }
#pragma unroll
    for (int j = 0; j < 4; ++j)
#pragma unroll
      for (int i = 0; i < 8; ++i)
        sm[(m0 + i) * 72 + j * 16 + col] = (_Float16)reluf(acc1[j][i] + bvu1[j]);
    LDS_FENCE();
    v16h h0 = a_load_h(sm, 72), h1 = a_load_h(sm + 32, 72);

    v16h w2[4][2];
#pragma unroll
    for (int j = 0; j < 4; ++j) {
      w2[j][0] = b_load(wq + WQ_U2, 64, j * 16, 0);
      w2[j][1] = b_load(wq + WQ_U2, 64, j * 16, 32);
    }
    v8f acc2[4];
#pragma unroll
    for (int j = 0; j < 4; ++j) { v8f z = {}; acc2[j] = wmma16(h0, w2[j][0], z); }
#pragma unroll
    for (int j = 0; j < 4; ++j) acc2[j] = wmma16(h1, w2[j][1], acc2[j]);
#pragma unroll
    for (int j = 0; j < 4; ++j) {
#pragma unroll
      for (int i = 0; i < 8; ++i) acc2[j][i] = reluf(acc2[j][i] + bvu2[j]);
      c_store_h(uemb + r0 * 64 + j * 16, 64, acc2[j]);  // row-major (A-load later)
    }
  }
}

// ---------------- persistent GRU scan: one wave per 16 batch rows ------------
__global__ __launch_bounds__(32) void gru_kernel(
    const _Float16* __restrict__ proj,   // frag-native [L*NRB][12][32][8]
    const _Float16* __restrict__ Whh, const float* __restrict__ bhh,
    _Float16* __restrict__ dseq,         // frag-native [L*NRB][4][32][8] or null
    float* __restrict__ hseq, float* __restrict__ hn) {
  __shared__ __align__(16) _Float16 smh[16 * 72];      // h C->A transpose
  __shared__ __align__(16) _Float16 smp[2][12 * 256];  // TDM double buffer (2 x 6KB)
  unsigned rb = blockIdx.x;
  size_t r0 = (size_t)rb * 16u;
  unsigned m0 = c_m0(), col = c_col();

  v16h wb[12][2];
#pragma unroll
  for (int j = 0; j < 12; ++j) {
    wb[j][0] = b_load(Whh, 64, j * 16, 0);
    wb[j][1] = b_load(Whh, 64, j * 16, 32);
  }
  float bj[12];
#pragma unroll
  for (int j = 0; j < 12; ++j) bj[j] = bhh[j * 16 + col];

  v8f hc[4];
#pragma unroll
  for (int c = 0; c < 4; ++c)
#pragma unroll
    for (int i = 0; i < 8; ++i) hc[c][i] = 0.f;
  v16h hA[2];
#pragma unroll
  for (int i = 0; i < 16; ++i) { hA[0][i] = (_Float16)0.f; hA[1][i] = (_Float16)0.f; }

  // prime the pipeline: DMA step-0 projection block (6144 B = 768 x 8B)
  tdm_fetch(proj + (size_t)rb * 3072, &smp[0][0], 768u);

  for (int t = 0; t < L_SEQ; ++t) {
    if (t + 1 < L_SEQ) {
      tdm_fetch(proj + ((size_t)(t + 1) * NRB + rb) * 3072, &smp[(t + 1) & 1][0], 768u);
      __builtin_amdgcn_s_wait_tensorcnt(1);  // in-order: step t's tile landed
    } else {
      __builtin_amdgcn_s_wait_tensorcnt(0);
    }
    const _Float16* pb = &smp[t & 1][0];

    if (hseq) {  // generative GRU records state *before* consuming step t
      float* hp = hseq + ((size_t)t * B_SZ + r0) * 64;
#pragma unroll
      for (int c = 0; c < 4; ++c) c_store_f(hp + c * 16, 64, hc[c]);
    }

    // gh = h @ Whh^T : two k-passes so dependent WMMAs sit 12 apart
    v8f acc[12];
#pragma unroll
    for (int j = 0; j < 12; ++j) { v8f z = {}; acc[j] = wmma16(hA[0], wb[j][0], z); }
#pragma unroll
    for (int j = 0; j < 12; ++j) acc[j] = wmma16(hA[1], wb[j][1], acc[j]);

#pragma unroll
    for (int c = 0; c < 4; ++c) {
      v8f xr = cn_load(pb + c * 256);            // ds_load_b128 each
      v8f xz = cn_load(pb + (4 + c) * 256);
      v8f xn = cn_load(pb + (8 + c) * 256);
#pragma unroll
      for (int i = 0; i < 8; ++i) {
        float r  = sigf(xr[i] + bj[c] + acc[c][i]);
        float zg = sigf(xz[i] + bj[4 + c] + acc[4 + c][i]);
        float nn = tanhf(xn[i] + bj[8 + c] + r * acc[8 + c][i]);
        float h  = (1.f - zg) * nn + zg * hc[c][i];
        hc[c][i] = h;
        smh[(m0 + i) * 72 + c * 16 + col] = (_Float16)h;
      }
    }
    if (dseq) {
      _Float16* dp = dseq + ((size_t)t * NRB + rb) * 1024;
#pragma unroll
      for (int c = 0; c < 4; ++c) cn_store(dp + c * 256, hc[c]);  // b128/lane
    }
    LDS_FENCE();
    hA[0] = a_load_h(smh, 72);
    hA[1] = a_load_h(smh + 32, 72);
  }
  if (hn) {
    float* hp = hn + r0 * 64;
#pragma unroll
    for (int c = 0; c < 4; ++c) c_store_f(hp + c * 16, 64, hc[c]);
  }
}

// ------- fused posterior DBlock + sample + z-embed + gen projection ----------
__global__ __launch_bounds__(256) void post_kernel(
    const _Float16* __restrict__ dseq, const float* __restrict__ eps,
    const _Float16* __restrict__ wq,
    const float* __restrict__ b1, const float* __restrict__ b2,
    const float* __restrict__ bmu, const float* __restrict__ bls,
    const float* __restrict__ bz1, const float* __restrict__ bz2,
    const float* __restrict__ bgih, const _Float16* __restrict__ uemb,
    float* __restrict__ zmu, float* __restrict__ zls, float* __restrict__ zout,
    _Float16* __restrict__ gproj) {
  __shared__ __align__(16) _Float16 lds[8][16 * 136];
  unsigned wid = threadIdx.x >> 5;
  _Float16* sm = lds[wid];
  size_t rbg = (size_t)blockIdx.x * 8u + wid;
  size_t r0 = rbg * 16u;
  unsigned m0 = c_m0(), col = c_col();

  float b1v[8], b2v[8], bmuv[2], blsv[2], bz1v[4], bz2v[4], bgv[12];
#pragma unroll
  for (int j = 0; j < 8; ++j) { b1v[j] = b1[j * 16 + col]; b2v[j] = b2[j * 16 + col]; }
#pragma unroll
  for (int n = 0; n < 2; ++n) { bmuv[n] = bmu[n * 16 + col]; blsv[n] = bls[n * 16 + col]; }
#pragma unroll
  for (int j = 0; j < 4; ++j) { bz1v[j] = bz1[j * 16 + col]; bz2v[j] = bz2[j * 16 + col]; }
#pragma unroll
  for (int j = 0; j < 12; ++j) bgv[j] = bgih[j * 16 + col];

  // re-stage fragment-native d tiles into row-major LDS, load as A-fragments
  {
    const _Float16* db = dseq + rbg * 1024;
#pragma unroll
    for (int c = 0; c < 4; ++c) {
      v8h h = cn_load_raw(db + c * 256);
#pragma unroll
      for (int i = 0; i < 8; ++i)
        sm[(m0 + i) * 72 + c * 16 + col] = h[i];
    }
  }
  LDS_FENCE();
  v16h d0 = a_load_h(sm, 72), d1 = a_load_h(sm + 32, 72);

  // DBlock hidden: t = tanh(d W1^T + b1) * sigmoid(d W2^T + b2)  [16 x 128]
  v8f g1a[8], g2a[8];
#pragma unroll
  for (int g = 0; g < 4; ++g) {   // groups of 2 tiles: 8 frags, 8 wmmas
    v16h wa[2][2], wbv[2][2];
#pragma unroll
    for (int q = 0; q < 2; ++q) {
      int j = g * 2 + q;
      wa[q][0]  = b_load(wq + WQ_P1, 64, j * 16, 0);
      wa[q][1]  = b_load(wq + WQ_P1, 64, j * 16, 32);
      wbv[q][0] = b_load(wq + WQ_P2, 64, j * 16, 0);
      wbv[q][1] = b_load(wq + WQ_P2, 64, j * 16, 32);
    }
#pragma unroll
    for (int q = 0; q < 2; ++q) {
      int j = g * 2 + q;
      v8f z1 = {}, z2 = {};
      z1 = wmma16(d0, wa[q][0], z1);
      z2 = wmma16(d0, wbv[q][0], z2);
      g1a[j] = wmma16(d1, wa[q][1], z1);
      g2a[j] = wmma16(d1, wbv[q][1], z2);
    }
  }
#pragma unroll
  for (int j = 0; j < 8; ++j)
#pragma unroll
    for (int i = 0; i < 8; ++i)
      sm[(m0 + i) * 136 + j * 16 + col] =
          (_Float16)(tanhf(g1a[j][i] + b1v[j]) * sigf(g2a[j][i] + b2v[j]));
  LDS_FENCE();
  v16h tA[4];
#pragma unroll
  for (int kc = 0; kc < 4; ++kc) tA[kc] = a_load_h(sm + kc * 32, 136);

  // mu / logsigma heads + reparameterized sample z = mu + exp(ls)*eps
#pragma unroll
  for (int n = 0; n < 2; ++n) {
    v8f cm = {}, cl = {};
#pragma unroll
    for (int kc = 0; kc < 4; ++kc) {
      cm = wmma16(tA[kc], b_load(wq + WQ_PMU, 128, n * 16, kc * 32), cm);
      cl = wmma16(tA[kc], b_load(wq + WQ_PLS, 128, n * 16, kc * 32), cl);
    }
    v8f ep = c_load_f(eps + r0 * 32 + n * 16, 32);
    v8f zz;
#pragma unroll
    for (int i = 0; i < 8; ++i) {
      cm[i] += bmuv[n];
      cl[i] += blsv[n];
      zz[i] = cm[i] + __expf(cl[i]) * ep[i];
      sm[(m0 + i) * 40 + n * 16 + col] = (_Float16)zz[i];
    }
    c_store_f(zmu + r0 * 32 + n * 16, 32, cm);
    c_store_f(zls + r0 * 32 + n * 16, 32, cl);
    c_store_f(zout + r0 * 32 + n * 16, 32, zz);
  }
  LDS_FENCE();
  v16h zA = a_load_h(sm, 40);

  // pz preprocess: 32 -> 64 (relu) -> 64 (relu) = z_embed
  v16h wz1[4];
#pragma unroll
  for (int j = 0; j < 4; ++j) wz1[j] = b_load(wq + WQ_Z1, 32, j * 16, 0);
  v8f p1a[4];
#pragma unroll
  for (int j = 0; j < 4; ++j) { v8f z = {}; p1a[j] = wmma16(zA, wz1[j], z); }
#pragma unroll
  for (int j = 0; j < 4; ++j)
#pragma unroll
    for (int i = 0; i < 8; ++i)
      sm[(m0 + i) * 72 + j * 16 + col] = (_Float16)reluf(p1a[j][i] + bz1v[j]);
  LDS_FENCE();
  v16h p0 = a_load_h(sm, 72), p1 = a_load_h(sm + 32, 72);
  v16h wz2[4][2];
#pragma unroll
  for (int j = 0; j < 4; ++j) {
    wz2[j][0] = b_load(wq + WQ_Z2, 64, j * 16, 0);
    wz2[j][1] = b_load(wq + WQ_Z2, 64, j * 16, 32);
  }
  v8f p2a[4];
#pragma unroll
  for (int j = 0; j < 4; ++j) { v8f z = {}; p2a[j] = wmma16(p0, wz2[j][0], z); }
#pragma unroll
  for (int j = 0; j < 4; ++j) p2a[j] = wmma16(p1, wz2[j][1], p2a[j]);
#pragma unroll
  for (int j = 0; j < 4; ++j)
#pragma unroll
    for (int i = 0; i < 8; ++i)
      sm[(m0 + i) * 72 + j * 16 + col] = (_Float16)reluf(p2a[j][i] + bz2v[j]);
  LDS_FENCE();
  v16h ze0 = a_load_h(sm, 72), ze1 = a_load_h(sm + 32, 72);
  v16h ue0 = a_load_h(uemb + r0 * 64, 64);
  v16h ue1 = a_load_h(uemb + r0 * 64 + 32, 64);

  // generative input projection: [z_embed | u_embed] (K=128) @ gen_Wih^T -> 192
#pragma unroll
  for (int g = 0; g < 6; ++g) {   // groups of 2 tiles: 8 frags, 8 wmmas
    v16h gw[2][4];
#pragma unroll
    for (int q = 0; q < 2; ++q) {
      int j = g * 2 + q;
      gw[q][0] = b_load(wq + WQ_GIH, 128, j * 16, 0);
      gw[q][1] = b_load(wq + WQ_GIH, 128, j * 16, 32);
      gw[q][2] = b_load(wq + WQ_GIH, 128, j * 16, 64);
      gw[q][3] = b_load(wq + WQ_GIH, 128, j * 16, 96);
    }
    v8f acc[2];
#pragma unroll
    for (int q = 0; q < 2; ++q) { v8f z = {}; acc[q] = wmma16(ze0, gw[q][0], z); }
#pragma unroll
    for (int q = 0; q < 2; ++q) acc[q] = wmma16(ze1, gw[q][1], acc[q]);
#pragma unroll
    for (int q = 0; q < 2; ++q) acc[q] = wmma16(ue0, gw[q][2], acc[q]);
#pragma unroll
    for (int q = 0; q < 2; ++q) acc[q] = wmma16(ue1, gw[q][3], acc[q]);
#pragma unroll
    for (int q = 0; q < 2; ++q)
      cn_store_b(gproj + rbg * 3072 + (g * 2 + q) * 256, acc[q], bgv[g * 2 + q]);
  }
}

extern "C" void kernel_launch(void* const* d_in, const int* in_sizes, int n_in,
                              void* d_out, int out_size, void* d_ws, size_t ws_size,
                              hipStream_t stream) {
  (void)in_sizes; (void)n_in; (void)out_size; (void)ws_size;
  const float* u_in  = (const float*)d_in[0];
  const float* x_in  = (const float*)d_in[1];
  const float* eps   = (const float*)d_in[2];
  const float* pu_W1 = (const float*)d_in[3];
  const float* pu_b1 = (const float*)d_in[4];
  const float* pu_W2 = (const float*)d_in[5];
  const float* pu_b2 = (const float*)d_in[6];
  const float* px_W1 = (const float*)d_in[7];
  const float* px_b1 = (const float*)d_in[8];
  const float* px_W2 = (const float*)d_in[9];
  const float* px_b2 = (const float*)d_in[10];
  const float* pz_W1 = (const float*)d_in[11];
  const float* pz_b1 = (const float*)d_in[12];
  const float* pz_W2 = (const float*)d_in[13];
  const float* pz_b2 = (const float*)d_in[14];
  const float* inf_Wih = (const float*)d_in[15];
  const float* inf_bih = (const float*)d_in[16];
  const float* inf_Whh = (const float*)d_in[17];
  const float* inf_bhh = (const float*)d_in[18];
  const float* gen_Wih = (const float*)d_in[19];
  const float* gen_bih = (const float*)d_in[20];
  const float* gen_Whh = (const float*)d_in[21];
  const float* gen_bhh = (const float*)d_in[22];
  const float* post_W1 = (const float*)d_in[23];
  const float* post_b1 = (const float*)d_in[24];
  const float* post_W2 = (const float*)d_in[25];
  const float* post_b2 = (const float*)d_in[26];
  const float* post_Wmu = (const float*)d_in[27];
  const float* post_bmu = (const float*)d_in[28];
  const float* post_Wls = (const float*)d_in[29];
  const float* post_bls = (const float*)d_in[30];

  _Float16* wq   = (_Float16*)d_ws;
  _Float16* proj = (_Float16*)((char*)d_ws + (256u << 10));
  _Float16* uemb = proj + (size_t)LB * 192;
  _Float16* dsq  = uemb + (size_t)LB * 64;

  float* out  = (float*)d_out;
  float* o_mu = out;
  float* o_ls = out + (size_t)LB * 32;
  float* o_z  = out + (size_t)LB * 64;
  float* o_h  = out + (size_t)LB * 96;
  float* o_xe = out + (size_t)LB * 160;
  float* o_hn = out + (size_t)LB * 224;

  PrepArgs pa;
  pa.seg[0]  = PrepSeg{px_W1, 2048, WQ_X1};
  pa.seg[1]  = PrepSeg{px_W2, 4096, WQ_X2};
  pa.seg[2]  = PrepSeg{pu_W1, 2048, WQ_U1};
  pa.seg[3]  = PrepSeg{pu_W2, 4096, WQ_U2};
  pa.seg[4]  = PrepSeg{pz_W1, 2048, WQ_Z1};
  pa.seg[5]  = PrepSeg{pz_W2, 4096, WQ_Z2};
  pa.seg[6]  = PrepSeg{inf_Wih, 12288, WQ_IIH};
  pa.seg[7]  = PrepSeg{inf_Whh, 12288, WQ_IHH};
  pa.seg[8]  = PrepSeg{post_W1, 8192, WQ_P1};
  pa.seg[9]  = PrepSeg{post_W2, 8192, WQ_P2};
  pa.seg[10] = PrepSeg{post_Wmu, 4096, WQ_PMU};
  pa.seg[11] = PrepSeg{post_Wls, 4096, WQ_PLS};
  pa.seg[12] = PrepSeg{gen_Wih, 24576, WQ_GIH};
  pa.seg[13] = PrepSeg{gen_Whh, 12288, WQ_GHH};
  prep_kernel<<<32, 256, 0, stream>>>(pa, wq);

  embed_kernel<<<LB / 16 / 8, 256, 0, stream>>>(
      x_in, u_in, wq, px_b1, px_b2, pu_b1, pu_b2, inf_bih, o_xe, uemb, proj);

  gru_kernel<<<NRB, 32, 0, stream>>>(proj, wq + WQ_IHH, inf_bhh,
                                     dsq, nullptr, nullptr);

  post_kernel<<<LB / 16 / 8, 256, 0, stream>>>(
      dsq, eps, wq, post_b1, post_b2, post_bmu, post_bls, pz_b1, pz_b2,
      gen_bih, uemb, o_mu, o_ls, o_z, proj);

  gru_kernel<<<NRB, 32, 0, stream>>>(proj, wq + WQ_GHH, gen_bhh,
                                     nullptr, o_h, o_hn);
}
